// Attention_63745904608049
// MI455X (gfx1250) — matrix operand order repaired and checked by reading, compiled, not run on hardware
//
#include <hip/hip_runtime.h>

// ---------------------------------------------------------------------------
// Relative-position attention (Transformer-XL / Enformer style) for MI455X.
// All matmuls use V_WMMA_F32_16X16X32_BF16 (wave32, f32 accumulate).
// v3: register ping-pong double buffering in both the GEMM K-loop and the
// attention J-loop so s_wait_loadcnt lands a full iteration after issue;
// near-scope prefetch.  Fragments remain contiguous 16/32-byte loads.
// ---------------------------------------------------------------------------

#define SEQ_LEN 2048
#define QDIM    1536
#define NHEAD   8
#define HDIM    64
#define INNER   512          // NHEAD * HDIM
#define PLEN    4095         // 2*SEQ-1
#define NB      256          // QDIM / 6

typedef __attribute__((ext_vector_type(16))) __bf16 v16bf;
typedef __attribute__((ext_vector_type(8)))  __bf16 v8bf;
typedef __attribute__((ext_vector_type(8)))  float  v8f;

__device__ __forceinline__ unsigned short f32_to_bf16(float f) {
  union { float f; unsigned int u; } x;
  x.f = f;
  unsigned int u = x.u;
  u += 0x7FFFu + ((u >> 16) & 1u);   // round-to-nearest-even
  return (unsigned short)(u >> 16);
}

__device__ __forceinline__ v8f v8f_zero() {
  v8f z;
#pragma unroll
  for (int i = 0; i < 8; i++) z[i] = 0.0f;
  return z;
}

// Merge two 16-byte halves into one A fragment (register-only).
__device__ __forceinline__ v16bf frag2(v8bf a, v8bf b) {
  return __builtin_shufflevector(a, b, 0, 1, 2, 3, 4, 5, 6, 7,
                                 8, 9, 10, 11, 12, 13, 14, 15);
}

// ---------------------------------------------------------------------------
// 1) casts: plain (row-major) and transposing (KxN f32 -> NxK bf16)
// ---------------------------------------------------------------------------
__global__ void cast_f32_to_bf16(const float* __restrict__ in,
                                 unsigned short* __restrict__ out, long n) {
  long i = (long)blockIdx.x * blockDim.x + threadIdx.x;
  if (i < n) out[i] = f32_to_bf16(in[i]);
}

__global__ void transpose_cast_f32_bf16(const float* __restrict__ in,
                                        unsigned short* __restrict__ out,
                                        int K, int N) {
  long i = (long)blockIdx.x * blockDim.x + threadIdx.x;
  if (i >= (long)K * N) return;
  int k = (int)(i / N), n = (int)(i % N);
  out[(long)n * K + k] = f32_to_bf16(in[i]);
}

// ---------------------------------------------------------------------------
// 2) Positional embedding (4095 x 1536) written directly as bf16.
// ---------------------------------------------------------------------------
__global__ __launch_bounds__(256) void pos_embed_kernel(unsigned short* __restrict__ emb) {
  __shared__ float red[256];
  int row = blockIdx.x;            // 0..4094
  int j   = threadIdx.x;           // 0..255
  float dist  = (float)(row - (SEQ_LEN - 1));
  float adist = fabsf(dist);
  float sgn   = (dist > 0.f) ? 1.f : (dist < 0.f ? -1.f : 0.f);

  float hl = exp2f(3.f + 8.f * (float)j / 255.f);     // 2^linspace(3,11,256)
  float ef = exp2f(-adist / hl);

  float cw = exp2f((float)(j + 1)) - 1.f;
  float cm = (cw > adist) ? 1.f : 0.f;

  float mean = 8.f + (2048.f - 8.f) * (float)j / 255.f;
  float conc = (mean * 0.25f) * (mean * 0.25f);
  float rate = mean * (1.f / 16.f);
  float log_unnorm = (conc - 1.f) * logf(adist) - rate * adist;
  float log_norm   = lgammaf(conc) - conc * logf(rate);
  float prob = expf(log_unnorm - log_norm) + 1e-8f;

  red[j] = prob;
  __syncthreads();
#pragma unroll
  for (int s = 128; s > 0; s >>= 1) {
    if (j < s) red[j] = fmaxf(red[j], red[j + s]);
    __syncthreads();
  }
  float gf = prob / red[0];

  long base = (long)row * QDIM;
  emb[base + 0 * NB + j] = f32_to_bf16(ef);
  emb[base + 1 * NB + j] = f32_to_bf16(cm);
  emb[base + 2 * NB + j] = f32_to_bf16(gf);
  emb[base + 3 * NB + j] = f32_to_bf16(sgn * ef);
  emb[base + 4 * NB + j] = f32_to_bf16(sgn * cm);
  emb[base + 5 * NB + j] = f32_to_bf16(sgn * gf);
}

// ---------------------------------------------------------------------------
// 3) LDS-free WMMA GEMM: C(f32, MxN) = A(MxK bf16 rowmaj) x BT(NxK bf16)ᵀ
//    [+ bias].  Block = 4 waves; wave tile = 32x64.  K-loop unrolled by 64
//    with ping-pong fragment buffers: next step's loads issue before the
//    current step's WMMAs, hiding L2 latency.
// ---------------------------------------------------------------------------
__global__ __launch_bounds__(128) void gemm_bf16_wmma(
    const unsigned short* __restrict__ A_, const unsigned short* __restrict__ BT_,
    const float* __restrict__ bias, float* __restrict__ C,
    int M, int N, int K) {
  const __bf16* A  = (const __bf16*)A_;
  const __bf16* BT = (const __bf16*)BT_;

  int tid  = threadIdx.x;
  int lane = tid & 31;
  int wave = tid >> 5;
  int ln   = lane & 15;
  int lh   = lane >> 4;
  int row0 = blockIdx.y * 128 + wave * 32;
  int col0 = blockIdx.x * 64;

  int ra0 = row0 + ln;      if (ra0 > M - 1) ra0 = M - 1;
  int ra1 = row0 + 16 + ln; if (ra1 > M - 1) ra1 = M - 1;
  const __bf16* arow0 = A + (long)ra0 * K;
  const __bf16* arow1 = A + (long)ra1 * K;
  const __bf16* brow[4];
#pragma unroll
  for (int cb = 0; cb < 4; cb++)
    brow[cb] = BT + (long)(col0 + cb * 16 + ln) * K;

  v8f acc[2][4];
#pragma unroll
  for (int mi = 0; mi < 2; mi++)
#pragma unroll
    for (int cb = 0; cb < 4; cb++) acc[mi][cb] = v8f_zero();

  auto loadA = [&](const __bf16* row, int k) -> v16bf {
    return frag2(*(const v8bf*)(row + k + lh * 16),
                 *(const v8bf*)(row + k + 8 + lh * 16));
  };

  // ping-pong fragment buffers (K is a multiple of 64 for all call sites)
  v16bf a0A, a1A, bA[4], a0B, a1B, bB[4];
  a0A = loadA(arow0, 0);
  a1A = loadA(arow1, 0);
#pragma unroll
  for (int cb = 0; cb < 4; cb++) bA[cb] = *(const v16bf*)(brow[cb] + lh * 16);

  for (int k0 = 0; k0 < K; k0 += 64) {
    // preload buffer B for k0+32 (always valid)
    a0B = loadA(arow0, k0 + 32);
    a1B = loadA(arow1, k0 + 32);
#pragma unroll
    for (int cb = 0; cb < 4; cb++)
      bB[cb] = *(const v16bf*)(brow[cb] + k0 + 32 + lh * 16);
    if (k0 + 64 < K) {
      __builtin_prefetch(arow0 + k0 + 128, 0, 3);
      __builtin_prefetch(brow[lane & 3] + k0 + 128, 0, 3);
    }
#pragma unroll
    for (int cb = 0; cb < 4; cb++) {
      acc[0][cb] = __builtin_amdgcn_wmma_f32_16x16x32_bf16(
          false, a0A, false, bA[cb], (short)0, acc[0][cb], false, false);
      acc[1][cb] = __builtin_amdgcn_wmma_f32_16x16x32_bf16(
          false, a1A, false, bA[cb], (short)0, acc[1][cb], false, false);
    }
    // preload buffer A for k0+64
    if (k0 + 64 < K) {
      a0A = loadA(arow0, k0 + 64);
      a1A = loadA(arow1, k0 + 64);
#pragma unroll
      for (int cb = 0; cb < 4; cb++)
        bA[cb] = *(const v16bf*)(brow[cb] + k0 + 64 + lh * 16);
    }
#pragma unroll
    for (int cb = 0; cb < 4; cb++) {
      acc[0][cb] = __builtin_amdgcn_wmma_f32_16x16x32_bf16(
          false, a0B, false, bB[cb], (short)0, acc[0][cb], false, false);
      acc[1][cb] = __builtin_amdgcn_wmma_f32_16x16x32_bf16(
          false, a1B, false, bB[cb], (short)0, acc[1][cb], false, false);
    }
  }

#pragma unroll
  for (int mi = 0; mi < 2; mi++)
#pragma unroll
    for (int cb = 0; cb < 4; cb++) {
      int col = col0 + cb * 16 + ln;
      float bv = bias ? bias[col] : 0.0f;
#pragma unroll
      for (int r = 0; r < 8; r++) {
        int row = row0 + mi * 16 + r + 8 * lh;   // C layout: VGPR r -> M=r+8*lh
        if (row < M) C[(long)row * N + col] = acc[mi][cb][r] + bv;
      }
    }
}

// ---------------------------------------------------------------------------
// 4) Head-major bf16 operands for fused attention.
// ---------------------------------------------------------------------------
__global__ void prep_qkv(const float* __restrict__ qf, const float* __restrict__ kf,
                         const float* __restrict__ vf, const float* __restrict__ cbias,
                         const float* __restrict__ pbias,
                         unsigned short* __restrict__ qc, unsigned short* __restrict__ qp,
                         unsigned short* __restrict__ kb, unsigned short* __restrict__ vt) {
  int i = blockIdx.x * blockDim.x + threadIdx.x;   // over SEQ*INNER
  if (i >= SEQ_LEN * INNER) return;
  int n = i / INNER, c = i % INNER;                // c = h*64 + d
  int h = c >> 6, d = c & 63;
  long dst = ((long)h * SEQ_LEN + n) * HDIM + d;
  const float scale = 0.125f;                      // HDIM^-0.5
  float q = qf[i] * scale;
  qc[dst] = f32_to_bf16(q + cbias[c]);
  qp[dst] = f32_to_bf16(q + pbias[c]);
  kb[dst] = f32_to_bf16(kf[i]);
  vt[((long)h * HDIM + d) * SEQ_LEN + n] = f32_to_bf16(vf[i]);
}

__global__ void prep_pos(const float* __restrict__ posf, unsigned short* __restrict__ pbm) {
  int i = blockIdx.x * blockDim.x + threadIdx.x;   // over PLEN*INNER
  if (i >= PLEN * INNER) return;
  int p = i / INNER, c = i % INNER;
  int h = c >> 6, d = c & 63;
  pbm[((long)h * PLEN + p) * HDIM + d] = f32_to_bf16(posf[i]);
}

// ---------------------------------------------------------------------------
// 5) Fused flash attention with relative shift.  1 wave per (head, 16-query
//    tile); per 32-column tile: 4 WMMAs qk + 6 WMMAs pos band + 4 WMMAs PV.
//    J-loop unrolled by 64 with ping-pong K/P/V fragment buffers.
// ---------------------------------------------------------------------------
__global__ __launch_bounds__(32) void attn_kernel(
    const unsigned short* __restrict__ qc_, const unsigned short* __restrict__ qp_,
    const unsigned short* __restrict__ kb_, const unsigned short* __restrict__ vt_,
    const unsigned short* __restrict__ pb_, float* __restrict__ O) {
  __shared__ float          band[16][48];
  __shared__ unsigned short pl[16][32];

  int lane = threadIdx.x;
  int ln = lane & 15, lh = lane >> 4;
  int h  = blockIdx.y;
  int i0 = blockIdx.x * 16;

  const __bf16* qc = (const __bf16*)qc_ + (long)h * SEQ_LEN * HDIM;
  const __bf16* qp = (const __bf16*)qp_ + (long)h * SEQ_LEN * HDIM;
  const __bf16* kb = (const __bf16*)kb_ + (long)h * SEQ_LEN * HDIM;
  const __bf16* vt = (const __bf16*)vt_ + (long)h * HDIM * SEQ_LEN;
  const __bf16* pm = (const __bf16*)pb_ + (long)h * PLEN * HDIM;

  // Q fragments (16x64 -> two 16x32 A-frags each)
  v16bf qca[2], qpa[2];
#pragma unroll
  for (int c = 0; c < 2; c++) {
    const __bf16* q0 = qc + (long)(i0 + ln) * HDIM + c * 32 + lh * 16;
    const __bf16* q1 = qp + (long)(i0 + ln) * HDIM + c * 32 + lh * 16;
    qca[c] = frag2(*(const v8bf*)q0, *(const v8bf*)(q0 + 8));
    qpa[c] = frag2(*(const v8bf*)q1, *(const v8bf*)(q1 + 8));
  }

  v8f o[4];
#pragma unroll
  for (int cb = 0; cb < 4; cb++) o[cb] = v8f_zero();
  float mrow[8], lrow[8];
#pragma unroll
  for (int r = 0; r < 8; r++) { mrow[r] = -1e30f; lrow[r] = 0.f; }

  auto loadK = [&](int j0, v16bf (&kf)[4]) {
#pragma unroll
    for (int sub = 0; sub < 2; sub++)
#pragma unroll
      for (int c = 0; c < 2; c++)
        kf[sub * 2 + c] = *(const v16bf*)(kb + (long)(j0 + sub * 16 + ln) * HDIM +
                                          c * 32 + lh * 16);
  };
  auto loadP = [&](int j0, v16bf (&pf)[6]) {
    int p0 = 2047 + j0 - i0 - 15;
#pragma unroll
    for (int nt = 0; nt < 3; nt++) {
      int prow = p0 + nt * 16 + ln;
      if (prow > PLEN - 1) prow = PLEN - 1;     // t=47 never consumed
#pragma unroll
      for (int c = 0; c < 2; c++)
        pf[nt * 2 + c] = *(const v16bf*)(pm + (long)prow * HDIM + c * 32 + lh * 16);
    }
  };
  auto loadV = [&](int j0, v16bf (&vv)[4]) {
#pragma unroll
    for (int cb = 0; cb < 4; cb++)
      vv[cb] = *(const v16bf*)(vt + (long)(cb * 16 + ln) * SEQ_LEN + j0 + lh * 16);
  };

  auto process = [&](int j0, v16bf (&kf)[4], v16bf (&pf)[6], v16bf (&vv)[4]) {
    // ---- qk scores: S[16x32] as two 16x16 subtiles
    v8f s[2];
#pragma unroll
    for (int sub = 0; sub < 2; sub++) {
      s[sub] = v8f_zero();
#pragma unroll
      for (int c = 0; c < 2; c++)
        s[sub] = __builtin_amdgcn_wmma_f32_16x16x32_bf16(
            false, qca[c], false, kf[sub * 2 + c], (short)0, s[sub], false, false);
    }
    // ---- positional band: R[m][t] = sum_d qp[i0+m,d] * P[p0+t, d]
    v8f rb[3];
#pragma unroll
    for (int nt = 0; nt < 3; nt++) {
      rb[nt] = v8f_zero();
#pragma unroll
      for (int c = 0; c < 2; c++)
        rb[nt] = __builtin_amdgcn_wmma_f32_16x16x32_bf16(
            false, qpa[c], false, pf[nt * 2 + c], (short)0, rb[nt], false, false);
    }

    __syncthreads();
#pragma unroll
    for (int nt = 0; nt < 3; nt++)
#pragma unroll
      for (int r = 0; r < 8; r++)
        band[r + 8 * lh][nt * 16 + ln] = rb[nt][r];
    __syncthreads();

    // ---- relative shift gather + online softmax
#pragma unroll
    for (int sub = 0; sub < 2; sub++)
#pragma unroll
      for (int r = 0; r < 8; r++) {
        int m = r + 8 * lh;
        int cc = sub * 16 + ln;
        s[sub][r] += band[m][cc - m + 15];
      }

    float pr0[8], pr1[8];
#pragma unroll
    for (int r = 0; r < 8; r++) {
      float tm = fmaxf(s[0][r], s[1][r]);
#pragma unroll
      for (int off = 1; off < 16; off <<= 1) tm = fmaxf(tm, __shfl_xor(tm, off, 16));
      float nm   = fmaxf(mrow[r], tm);
      float corr = expf(mrow[r] - nm);
      float p0v  = expf(s[0][r] - nm);
      float p1v  = expf(s[1][r] - nm);
      float ts   = p0v + p1v;
#pragma unroll
      for (int off = 1; off < 16; off <<= 1) ts += __shfl_xor(ts, off, 16);
      lrow[r] = lrow[r] * corr + ts;
      mrow[r] = nm;
#pragma unroll
      for (int cb = 0; cb < 4; cb++) o[cb][r] *= corr;
      pr0[r] = p0v; pr1[r] = p1v;
    }

    // ---- probs: C layout -> A layout via LDS
#pragma unroll
    for (int r = 0; r < 8; r++) {
      pl[r + 8 * lh][ln]      = f32_to_bf16(pr0[r]);
      pl[r + 8 * lh][16 + ln] = f32_to_bf16(pr1[r]);
    }
    __syncthreads();

    const __bf16* plb = (const __bf16*)&pl[0][0] + ln * 32 + lh * 16;
    v16bf pa = frag2(*(const v8bf*)plb, *(const v8bf*)(plb + 8));

    // ---- PV: o(16x64) += P(16x32) x V(32x64)
#pragma unroll
    for (int cb = 0; cb < 4; cb++)
      o[cb] = __builtin_amdgcn_wmma_f32_16x16x32_bf16(
          false, pa, false, vv[cb], (short)0, o[cb], false, false);
    __syncthreads();
  };

  // ping-pong over 32-column tiles (SEQ/32 = 64 tiles, even count)
  v16bf kfA[4], pfA[6], vfA[4], kfB[4], pfB[6], vfB[4];
  loadK(0, kfA); loadP(0, pfA); loadV(0, vfA);
  for (int j0 = 0; j0 < SEQ_LEN; j0 += 64) {
    loadK(j0 + 32, kfB); loadP(j0 + 32, pfB); loadV(j0 + 32, vfB);
    process(j0, kfA, pfA, vfA);
    if (j0 + 64 < SEQ_LEN) {
      loadK(j0 + 64, kfA); loadP(j0 + 64, pfA); loadV(j0 + 64, vfA);
    }
    process(j0 + 32, kfB, pfB, vfB);
  }

  // ---- normalize + write O[n][h*64+d]
#pragma unroll
  for (int cb = 0; cb < 4; cb++)
#pragma unroll
    for (int r = 0; r < 8; r++) {
      int m = r + 8 * lh;
      O[(long)(i0 + m) * INNER + h * HDIM + cb * 16 + ln] = o[cb][r] / lrow[r];
    }
}

// ---------------------------------------------------------------------------
// Launcher
// ---------------------------------------------------------------------------
extern "C" void kernel_launch(void* const* d_in, const int* in_sizes, int n_in,
                              void* d_out, int out_size, void* d_ws, size_t ws_size,
                              hipStream_t stream) {
  const float* x    = (const float*)d_in[0];
  const float* Wq   = (const float*)d_in[1];
  const float* Wk   = (const float*)d_in[2];
  const float* Wv   = (const float*)d_in[3];
  const float* cbias= (const float*)d_in[4];
  const float* pbias= (const float*)d_in[5];
  const float* Wpw  = (const float*)d_in[6];
  const float* Wpb  = (const float*)d_in[7];
  const float* Wow  = (const float*)d_in[8];
  const float* Wob  = (const float*)d_in[9];
  float* out = (float*)d_out;

  char* ws = (char*)d_ws;
  size_t off = 0;
  auto alloc = [&](size_t bytes) -> void* {
    size_t o = off;
    off += (bytes + 255) & ~(size_t)255;
    return (void*)(ws + o);
  };

  unsigned short* x_bf    = (unsigned short*)alloc((size_t)SEQ_LEN * QDIM * 2);
  unsigned short* wq_bt   = (unsigned short*)alloc((size_t)QDIM * INNER * 2);  // INNER x QDIM
  unsigned short* wk_bt   = (unsigned short*)alloc((size_t)QDIM * INNER * 2);
  unsigned short* wv_bt   = (unsigned short*)alloc((size_t)QDIM * INNER * 2);
  unsigned short* wp_bt   = (unsigned short*)alloc((size_t)QDIM * INNER * 2);
  unsigned short* wo_bt   = (unsigned short*)alloc((size_t)INNER * QDIM * 2);  // QDIM x INNER
  unsigned short* emb_bf  = (unsigned short*)alloc((size_t)PLEN * QDIM * 2);
  float*          qf      = (float*)alloc((size_t)SEQ_LEN * INNER * 4);
  float*          kf      = (float*)alloc((size_t)SEQ_LEN * INNER * 4);
  float*          vf      = (float*)alloc((size_t)SEQ_LEN * INNER * 4);
  float*          posf    = (float*)alloc((size_t)PLEN * INNER * 4);
  unsigned short* qc_bf   = (unsigned short*)alloc((size_t)SEQ_LEN * INNER * 2);
  unsigned short* qp_bf   = (unsigned short*)alloc((size_t)SEQ_LEN * INNER * 2);
  unsigned short* k_bf    = (unsigned short*)alloc((size_t)SEQ_LEN * INNER * 2);
  unsigned short* v_bt    = (unsigned short*)alloc((size_t)SEQ_LEN * INNER * 2);
  unsigned short* p_bf    = (unsigned short*)alloc((size_t)PLEN * INNER * 2);
  float*          attnO   = (float*)alloc((size_t)SEQ_LEN * INNER * 4);
  unsigned short* attn_bf = (unsigned short*)alloc((size_t)SEQ_LEN * INNER * 2);

  auto cast = [&](const float* src, unsigned short* dst, long n) {
    int blocks = (int)((n + 255) / 256);
    cast_f32_to_bf16<<<blocks, 256, 0, stream>>>(src, dst, n);
  };
  auto tcast = [&](const float* src, unsigned short* dst, int K, int N) {
    long n = (long)K * N;
    int blocks = (int)((n + 255) / 256);
    transpose_cast_f32_bf16<<<blocks, 256, 0, stream>>>(src, dst, K, N);
  };

  // 1) operands: A row-major, weights transposed (NxK)
  cast(x, x_bf, (long)SEQ_LEN * QDIM);
  tcast(Wq,  wq_bt, QDIM, INNER);
  tcast(Wk,  wk_bt, QDIM, INNER);
  tcast(Wv,  wv_bt, QDIM, INNER);
  tcast(Wpw, wp_bt, QDIM, INNER);
  tcast(Wow, wo_bt, INNER, QDIM);

  // 2) positional features
  pos_embed_kernel<<<PLEN, 256, 0, stream>>>(emb_bf);

  // 3) projections: q, k, v  (2048 x 512 x 1536)
  dim3 gq(INNER / 64, SEQ_LEN / 128);
  gemm_bf16_wmma<<<gq, 128, 0, stream>>>(x_bf, wq_bt, nullptr, qf, SEQ_LEN, INNER, QDIM);
  gemm_bf16_wmma<<<gq, 128, 0, stream>>>(x_bf, wk_bt, nullptr, kf, SEQ_LEN, INNER, QDIM);
  gemm_bf16_wmma<<<gq, 128, 0, stream>>>(x_bf, wv_bt, nullptr, vf, SEQ_LEN, INNER, QDIM);

  // 4) positional projection: (4095 x 512 x 1536) + Wp_b
  dim3 gp(INNER / 64, (PLEN + 127) / 128);
  gemm_bf16_wmma<<<gp, 128, 0, stream>>>(emb_bf, wp_bt, Wpb, posf, PLEN, INNER, QDIM);

  // 5) head-major bf16 operands (scale + biases folded into q; V transposed)
  prep_qkv<<<(SEQ_LEN * INNER) / 256, 256, 0, stream>>>(
      qf, kf, vf, cbias, pbias, qc_bf, qp_bf, k_bf, v_bt);
  prep_pos<<<(PLEN * INNER + 255) / 256, 256, 0, stream>>>(posf, p_bf);

  // 6) fused flash attention with relative shift
  dim3 ga(SEQ_LEN / 16, NHEAD);
  attn_kernel<<<ga, 32, 0, stream>>>(qc_bf, qp_bf, k_bf, v_bt, p_bf, attnO);

  // 7) output projection: (2048 x 1536 x 512) + Wout_b -> d_out
  cast(attnO, attn_bf, (long)SEQ_LEN * INNER);
  dim3 go(QDIM / 64, SEQ_LEN / 128);
  gemm_bf16_wmma<<<go, 128, 0, stream>>>(attn_bf, wo_bt, Wob, out, SEQ_LEN, QDIM, INNER);
}
